// NSATransformerEncoderLayer_82858509074963
// MI455X (gfx1250) — matrix operand order repaired and checked by reading, not compiled, not run on hardware
//
#include <hip/hip_runtime.h>
#include <cmath>

typedef __attribute__((ext_vector_type(16))) _Float16 v16h;
typedef __attribute__((ext_vector_type(8)))  _Float16 v8h;
typedef __attribute__((ext_vector_type(8)))  float    v8f;

#define DEV static __device__ __forceinline__

// ---------------- problem constants ----------------
constexpr int Bc = 4, Sc = 1024, DMc = 512, Hc = 8, Dc = 64;
constexpr int WCMP = 4, STRIDEc = 2, BSc = 4, DFFc = 2048;
constexpr int NCc = (Sc - WCMP) / STRIDEc + 1;   // 511 compressed blocks
constexpr int NKc = NCc + 1;                     // 512 keys incl. mem slot
constexpr float SCALE = 0.125f;                  // D^-0.5

// ---------------- WMMA fragment helpers (wave32, 16x16x32 f16) ----------------
// 16 contiguous halfs -> v16h (two 16-byte loads)
DEV v16h ld16(const _Float16* p) {
  v16h r;
  ((v8h*)&r)[0] = *(const v8h*)(p);
  ((v8h*)&r)[1] = *(const v8h*)(p + 8);
  return r;
}

// A fragment (M = lane%16): halves h -> K = (h&7) + 16*(h>>3) + 8*(lane>=16)
// == two contiguous 8-half runs from a row-major K-contiguous row.
DEV v16h loadA(const _Float16* rowp, int k0, int lane) {
  int hi = (lane >> 4) << 3;   // 0 or 8
  v16h r;
  ((v8h*)&r)[0] = *(const v8h*)(rowp + k0 + hi);
  ((v8h*)&r)[1] = *(const v8h*)(rowp + k0 + 16 + hi);
  return r;
}

// B fragment (N = lane%16): halves h -> K = (h&7) + 16*(h>>3) + 8*(lane>=16)
// == two contiguous 8-half runs from an N-major (transposed) K-contiguous row.
DEV v16h loadB(const _Float16* rowp, int k0, int lane) {
  int hi = (lane >> 4) << 3;
  v16h r;
  ((v8h*)&r)[0] = *(const v8h*)(rowp + k0 + hi);
  ((v8h*)&r)[1] = *(const v8h*)(rowp + k0 + 16 + hi);
  return r;
}

DEV float halfmax16(float v) {
  for (int m = 8; m > 0; m >>= 1) v = fmaxf(v, __shfl_xor(v, m, 32));
  return v;
}
DEV float halfsum16(float v) {
  for (int m = 8; m > 0; m >>= 1) v += __shfl_xor(v, m, 32);
  return v;
}

// ---------------- transpose + cast: in (K,N) f32 -> out (N,K) f16 ----------------
__global__ void transcast_kernel(const float* __restrict__ in, _Float16* __restrict__ out,
                                 int K, int N) {
  int idx = blockIdx.x * 256 + threadIdx.x;
  if (idx >= N * K) return;
  int n = idx / K, k = idx - n * K;
  out[idx] = (_Float16)in[(size_t)k * N + n];
}

// ---------------- LayerNorm (one 512-wide row per block of 128) ----------------
__global__ void ln_kernel(const float* __restrict__ x, const float* __restrict__ g,
                          const float* __restrict__ b, _Float16* __restrict__ out) {
  __shared__ float red[128];
  int row = blockIdx.x, t = threadIdx.x;
  const float* xr = x + (size_t)row * DMc;
  float v4[4], s = 0.f;
  for (int i = 0; i < 4; i++) { v4[i] = xr[t + i * 128]; s += v4[i]; }
  red[t] = s; __syncthreads();
  for (int st = 64; st > 0; st >>= 1) { if (t < st) red[t] += red[t + st]; __syncthreads(); }
  float mean = red[0] / DMc; __syncthreads();
  s = 0.f;
  for (int i = 0; i < 4; i++) { float d = v4[i] - mean; s += d * d; }
  red[t] = s; __syncthreads();
  for (int st = 64; st > 0; st >>= 1) { if (t < st) red[t] += red[t + st]; __syncthreads(); }
  float rstd = rsqrtf(red[0] / DMc + 1e-5f);
  for (int i = 0; i < 4; i++) {
    int c = t + i * 128;
    out[(size_t)row * DMc + c] = (_Float16)((v4[i] - mean) * rstd * g[c] + b[c]);
  }
}

// ---------------- WMMA GEMM, no LDS: C = A(M,K) @ BT(N,K)^T  [+bias][gelu][+resid] ----
// Compile-time N/K: row strides fold into immediate store offsets.
// block = 256 (8 waves); wave computes a 32x32 tile; block covers 128(M) x 64(N)
template<int OUTH, int GELU, int RESID, int N, int K>
__global__ void wmma_gemm_kernel(const _Float16* __restrict__ A, const _Float16* __restrict__ BT,
                                 const float* __restrict__ bias, const float* __restrict__ resid,
                                 void* __restrict__ Cout) {
  int lane = threadIdx.x & 31;
  int wave = threadIdx.x >> 5;
  int m0 = blockIdx.y * 128 + (wave & 3) * 32;
  int n0 = blockIdx.x * 64 + (wave >> 2) * 32;
  const _Float16* Ar0 = A + (size_t)(m0 + (lane & 15)) * K;
  const _Float16* Ar1 = Ar0 + (size_t)16 * K;
  const _Float16* Br0 = BT + (size_t)(n0 + (lane & 15)) * K;
  const _Float16* Br1 = Br0 + (size_t)16 * K;
  v8f acc00 = {}, acc01 = {}, acc10 = {}, acc11 = {};
  for (int k0 = 0; k0 < K; k0 += 32) {
    __builtin_prefetch(Ar0 + k0 + 256, 0, 1);
    __builtin_prefetch(Br0 + k0 + 256, 0, 1);
    v16h a0 = loadA(Ar0, k0, lane);
    v16h a1 = loadA(Ar1, k0, lane);
    v16h b0 = loadB(Br0, k0, lane);
    v16h b1 = loadB(Br1, k0, lane);
    acc00 = __builtin_amdgcn_wmma_f32_16x16x32_f16(false, a0, false, b0, (short)0, acc00, false, false);
    acc01 = __builtin_amdgcn_wmma_f32_16x16x32_f16(false, a0, false, b1, (short)0, acc01, false, false);
    acc10 = __builtin_amdgcn_wmma_f32_16x16x32_f16(false, a1, false, b0, (short)0, acc10, false, false);
    acc11 = __builtin_amdgcn_wmma_f32_16x16x32_f16(false, a1, false, b1, (short)0, acc11, false, false);
  }
  int hr = (lane >> 4) << 3;
  auto store_tile = [&](const v8f& acc, int mt, int nt) {
    int nn = n0 + nt * 16 + (lane & 15);
    float bv = bias ? bias[nn] : 0.f;
    size_t bofs = (size_t)(m0 + mt * 16 + hr) * N + nn;   // row stride N is constexpr
    const float* rp = RESID ? (resid + bofs) : nullptr;
    _Float16* oh = (_Float16*)Cout + bofs;
    float*    of = (float*)Cout + bofs;
#pragma unroll
    for (int r = 0; r < 8; r++) {
      float v = acc[r] + bv;
      if (GELU) v = 0.5f * v * (1.f + erff(v * 0.70710678118f));
      if (RESID) v += rp[r * N];
      if (OUTH) oh[r * N] = (_Float16)v;
      else      of[r * N] = v;
    }
  };
  store_tile(acc00, 0, 0); store_tile(acc01, 0, 1);
  store_tile(acc10, 1, 0); store_tile(acc11, 1, 1);
}

// ---------------- mem_ck / mem_cv -> slot 0 of compressed K / V^T ----------------
__global__ void memrow_kernel(const float* __restrict__ mck, const float* __restrict__ mcv,
                              _Float16* __restrict__ ck, _Float16* __restrict__ cvT) {
  int bh = blockIdx.x, d = threadIdx.x, h = bh % Hc;
  ck[(size_t)bh * NKc * Dc + d] = (_Float16)mck[h * Dc + d];           // key-major, key 0
  cvT[((size_t)bh * Dc + d) * NKc + 0] = (_Float16)mcv[h * Dc + d];    // d-major, key 0
}

// ---------------- ck (key-major) and cvT (d-major) via WMMA (K=256) ----------------
__global__ void ckcv_kernel(const _Float16* __restrict__ kh, const _Float16* __restrict__ vh,
                            const float* __restrict__ kpe, const float* __restrict__ vpe,
                            const _Float16* __restrict__ WckT, const _Float16* __restrict__ WcvT,
                            const float* __restrict__ bck, const float* __restrict__ bcv,
                            _Float16* __restrict__ ck, _Float16* __restrict__ cvT) {
  int lane = threadIdx.x & 31;
  int c0 = blockIdx.x * 16;
  int bh = blockIdx.y;
  int b = bh / Hc, h = bh % Hc;
  int doCV = blockIdx.z;
  const _Float16* srck = doCV ? vh : kh;
  const float*    pe   = doCV ? vpe : kpe;
  const _Float16* WT   = doCV ? WcvT : WckT;    // (64, 256) N-major
  const float*    bb   = doCV ? bcv : bck;

  int c = c0 + (lane & 15); if (c >= NCc) c = NCc - 1;    // clamp for A loads
  v8f acc[4];
  for (int t = 0; t < 4; t++) for (int r = 0; r < 8; r++) acc[t][r] = 0.f;
  int hi = (lane >> 4) << 3;
  for (int k0 = 0; k0 < 256; k0 += 32) {
    v16h a;
    for (int part = 0; part < 2; part++) {
      int kbase = k0 + hi + part * 16;                    // multiple of 8, no 64-boundary cross
      int w = kbase >> 6, d = kbase & 63;
      int s = STRIDEc * c + w;
      const _Float16* kr = srck + (((size_t)(b * Sc + s) * Hc + h) * Dc);
      v8h kv = *(const v8h*)(kr + d);
      const float* per = pe + (size_t)(h * WCMP + w) * Dc + d;
      for (int e = 0; e < 8; e++)
        a[part * 8 + e] = (_Float16)((float)kv[e] + per[e]);
    }
    for (int t = 0; t < 4; t++) {
      const _Float16* Wrow = WT + (size_t)(t * 16 + (lane & 15)) * 256;
      v16h bf = loadB(Wrow, k0, lane);
      acc[t] = __builtin_amdgcn_wmma_f32_16x16x32_f16(false, a, false, bf, (short)0, acc[t], false, false);
    }
  }
  int hr = (lane >> 4) << 3;
  bool full = (c0 + 16 <= NCc);                 // all rows of this tile valid?
  for (int t = 0; t < 4; t++) {
    int n = t * 16 + (lane & 15);
    float bvv = bb[n];
    if (doCV) {
      _Float16* p = cvT + ((size_t)bh * Dc + n) * NKc + 1 + c0 + hr;   // per-r stride 1
      if (full) {
#pragma unroll
        for (int r = 0; r < 8; r++) p[r] = (_Float16)(acc[t][r] + bvv);
      } else {
        for (int r = 0; r < 8; r++)
          if (c0 + hr + r < NCc) p[r] = (_Float16)(acc[t][r] + bvv);
      }
    } else {
      _Float16* p = ck + ((size_t)bh * NKc + 1 + c0 + hr) * Dc + n;    // per-r stride Dc
      if (full) {
#pragma unroll
        for (int r = 0; r < 8; r++) p[r * Dc] = (_Float16)(acc[t][r] + bvv);
      } else {
        for (int r = 0; r < 8; r++)
          if (c0 + hr + r < NCc) p[r * Dc] = (_Float16)(acc[t][r] + bvv);
      }
    }
  }
}

// ---------------- gate logits: sigmoid(xn @ Wg + bg), one wave per token ----------------
__global__ void gates_kernel(const _Float16* __restrict__ xnh, const float* __restrict__ Wg,
                             const float* __restrict__ bg, float* __restrict__ gates) {
  int row = blockIdx.x, lane = threadIdx.x;
  float acc[24];
  for (int j = 0; j < 24; j++) acc[j] = 0.f;
  const _Float16* xr = xnh + (size_t)row * DMc;
  for (int d = lane; d < DMc; d += 32) {
    float xv = (float)xr[d];
    const float* wr = Wg + (size_t)d * 24;
    for (int j = 0; j < 24; j++) acc[j] += xv * wr[j];
  }
  for (int m = 16; m > 0; m >>= 1)
    for (int j = 0; j < 24; j++) acc[j] += __shfl_xor(acc[j], m, 32);
  if (lane == 0)
    for (int j = 0; j < 24; j++)
      gates[(size_t)row * 24 + j] = 1.f / (1.f + __expf(-(acc[j] + bg[j])));
}

// ---------------- compressed attention (WMMA flash) + importance top-2 selection ----------------
#define CWAVES 4
DEV void score16(const v16h& aq0, const v16h& aq1, const _Float16* ckb, int jj0,
                 int lane, int s0, float* out8) {
  const _Float16* ckr = ckb + (size_t)(jj0 + (lane & 15)) * Dc;   // key row, d-contiguous
  v16h b0 = loadB(ckr, 0, lane);                                  // d 0..31 slice
  v16h b1 = loadB(ckr, 32, lane);                                 // d 32..63 slice
  v8f c8 = {};
  c8 = __builtin_amdgcn_wmma_f32_16x16x32_f16(false, aq0, false, b0, (short)0, c8, false, false);
  c8 = __builtin_amdgcn_wmma_f32_16x16x32_f16(false, aq1, false, b1, (short)0, c8, false, false);
  int jj = jj0 + (lane & 15);
  int hi8 = (lane >> 4) << 3;
#pragma unroll
  for (int r = 0; r < 8; r++) {
    int qpos = s0 + r + hi8;
    bool vis = (jj == 0) || (qpos >= 2 * jj + 1); // starts[c]+W_CMP-1 = 2(jj-1)+3
    out8[r] = vis ? c8[r] * SCALE : -1e30f;
  }
}

__global__ void cattn_kernel(const _Float16* __restrict__ qh,
                             const _Float16* __restrict__ ck,
                             const _Float16* __restrict__ cvT,
                             float* __restrict__ cout, int* __restrict__ sel) {
  __shared__ alignas(64) _Float16 Lp[CWAVES][16 * 32];
  int lane = threadIdx.x & 31;
  int wave = threadIdx.x >> 5;
  int bh = blockIdx.x;
  int b = bh >> 3, h = bh & 7;
  int s0 = (blockIdx.y * CWAVES + wave) * 16;
  const _Float16* qbase = qh + (((size_t)(b * Sc + s0 + (lane & 15)) * Hc + h) * Dc);
  v16h aq0 = loadA(qbase, 0, lane);
  v16h aq1 = loadA(qbase, 32, lane);
  const _Float16* ckb = ck + (size_t)bh * NKc * Dc;
  const _Float16* cvb = cvT + (size_t)bh * Dc * NKc;
  int hi8 = (lane >> 4) << 3;

  // ---- pass 1: exact row max / sum (online) ----
  float rm[8], rs[8];
  for (int r = 0; r < 8; r++) { rm[r] = -1e30f; rs[r] = 0.f; }
  for (int j0 = 0; j0 < NKc; j0 += 32) {
    float sc0[8], sc1[8];
    score16(aq0, aq1, ckb, j0, lane, s0, sc0);
    score16(aq0, aq1, ckb, j0 + 16, lane, s0, sc1);
    for (int r = 0; r < 8; r++) {
      float bm = halfmax16(fmaxf(sc0[r], sc1[r]));
      float mnew = fmaxf(rm[r], bm);
      float bs = halfsum16(__expf(sc0[r] - mnew) + __expf(sc1[r] - mnew));
      rs[r] = rs[r] * __expf(rm[r] - mnew) + bs;
      rm[r] = mnew;
    }
  }

  // ---- pass 2: probs, P·V via WMMA, importance scan + top-2 ----
  v8f oacc[4];
  for (int t = 0; t < 4; t++) for (int r = 0; r < 8; r++) oacc[t][r] = 0.f;
  float b1v = -1.f, b2v = -2.f; int b1i = 0, b2i = 0;
  int cur_ns = -1; float cur_sum = 0.f;
  int own = (s0 + lane) >> 2;                       // used by lanes < 16 (row m = lane)
  _Float16* lp = Lp[wave];

  for (int j0 = 0; j0 < NKc; j0 += 32) {
    for (int half = 0; half < 2; half++) {
      float scx[8];
      score16(aq0, aq1, ckb, j0 + half * 16, lane, s0, scx);
#pragma unroll
      for (int r = 0; r < 8; r++) {
        int m = r + hi8;
        float p = __expf(scx[r] - rm[r]);
        lp[m * 32 + half * 16 + (lane & 15)] = (_Float16)p;
      }
    }
    asm volatile("" ::: "memory");                  // keep LDS store->load order
    // A fragment of probs: standard row-major A layout -> two 16B LDS loads
    v16h ap = loadA(lp + (lane & 15) * 32, 0, lane);
    for (int t = 0; t < 4; t++) {
      // B fragment from d-major cvT: contiguous keys per d row
      const _Float16* cvrow = cvb + (size_t)(t * 16 + (lane & 15)) * NKc;
      v16h bv = loadB(cvrow, j0, lane);
      oacc[t] = __builtin_amdgcn_wmma_f32_16x16x32_f16(false, ap, false, bv, (short)0, oacc[t], false, false);
    }
    if (lane < 16) {                                // per-row importance scan over this tile
      for (int kk = 0; kk < 32; kk++) {
        int jj = j0 + kk;
        if (jj == 0) continue;
        int ns = (jj - 1) >> 1;
        if (ns != cur_ns) {
          if (cur_ns >= 0) {
            float val = (cur_ns == own) ? 1e9f : cur_sum;
            if (val > b1v) { b2v = b1v; b2i = b1i; b1v = val; b1i = cur_ns; }
            else if (val > b2v) { b2v = val; b2i = cur_ns; }
          }
          cur_ns = ns; cur_sum = 0.f;
        }
        cur_sum += (float)lp[lane * 32 + kk];
      }
    }
  }
  if (lane < 16 && cur_ns >= 0) {
    float val = (cur_ns == own) ? 1e9f : cur_sum;
    if (val > b1v) { b2v = b1v; b2i = b1i; b1v = val; b1i = cur_ns; }
    else if (val > b2v) { b2v = val; b2i = cur_ns; }
  }
  // cout stores: base + constexpr immediate offsets (row stride Dc)
  float* cbase = cout + ((size_t)bh * Sc + s0 + hi8) * Dc + (lane & 15);
#pragma unroll
  for (int t = 0; t < 4; t++)
#pragma unroll
    for (int r = 0; r < 8; r++)
      cbase[r * Dc + t * 16] = oacc[t][r] / rs[r];
  if (lane < 16) {
    size_t o = ((size_t)bh * Sc + s0 + lane) * 2;
    sel[o] = b1i; sel[o + 1] = b2i;
  }
}

// ---------------- fine (8 selected keys) + sliding (2 keys) attention + gating ----------------
// one wave per (b,h,s); cooperative scores: 4 lanes x 16 dims per fine key
__global__ void fsg_kernel(const _Float16* __restrict__ qh, const _Float16* __restrict__ kh,
                           const _Float16* __restrict__ vh, const int* __restrict__ sel,
                           const float* __restrict__ cout, const float* __restrict__ gates,
                           _Float16* __restrict__ oh) {
  int lane = threadIdx.x & 31;
  int wave = threadIdx.x >> 5;
  long idx = (long)blockIdx.x * 8 + wave;          // (b*H + h)*S + s
  int s = (int)(idx & (Sc - 1));
  int bh = (int)(idx >> 10);
  int b = bh >> 3, h = bh & 7;
  const _Float16* qr = qh + ((size_t)(b * Sc + s) * Hc + h) * Dc;
  int sel0 = sel[idx * 2], sel1 = sel[idx * 2 + 1];

  // --- fine: key t = lane>>2, dim chunk sub = lane&3 (16 dims each) ---
  int t = lane >> 2, sub = lane & 3;
  int blk = (t < 4) ? sel0 : sel1;
  int pos = blk * BSc + (t & 3);
  const _Float16* kr = kh + ((size_t)(b * Sc + pos) * Hc + h) * Dc;
  float part = 0.f;
  for (int dd = sub * 16; dd < sub * 16 + 16; dd++)
    part += (float)qr[dd] * (float)kr[dd];
  part += __shfl_xor(part, 1, 32);
  part += __shfl_xor(part, 2, 32);                  // full dot, replicated in quad
  float sc = (pos <= s) ? part * SCALE : -1e30f;
  float mx = sc;
  mx = fmaxf(mx, __shfl_xor(mx, 4, 32));
  mx = fmaxf(mx, __shfl_xor(mx, 8, 32));
  mx = fmaxf(mx, __shfl_xor(mx, 16, 32));          // max over 8 keys
  float pr = __expf(sc - mx);
  float sum = pr;
  sum += __shfl_xor(sum, 4, 32);
  sum += __shfl_xor(sum, 8, 32);
  sum += __shfl_xor(sum, 16, 32);
  float sinv = 1.f / sum;
  int d0 = lane * 2;
  float f0 = 0.f, f1 = 0.f;
  for (int t2 = 0; t2 < 8; t2++) {
    float w = __shfl(pr, t2 * 4, 32) * sinv;
    int pt = __shfl(pos, t2 * 4, 32);
    const _Float16* vr = vh + ((size_t)(b * Sc + pt) * Hc + h) * Dc;
    f0 += w * (float)vr[d0];
    f1 += w * (float)vr[d0 + 1];
  }

  // --- sliding: key half = lane>>4 (s-1 then s), 4 dims per lane ---
  int kidx = lane >> 4, sub16 = lane & 15;
  int sp0 = (s >= 1) ? s - 1 : 0;
  int sp = kidx ? s : sp0;
  const _Float16* skr = kh + ((size_t)(b * Sc + sp) * Hc + h) * Dc;
  float sp4 = 0.f;
  for (int dd = sub16 * 4; dd < sub16 * 4 + 4; dd++)
    sp4 += (float)qr[dd] * (float)skr[dd];
  sp4 += __shfl_xor(sp4, 1, 32);
  sp4 += __shfl_xor(sp4, 2, 32);
  sp4 += __shfl_xor(sp4, 4, 32);
  sp4 += __shfl_xor(sp4, 8, 32);                   // per-half full dot
  float w0 = __shfl(sp4, 0, 32);
  float w1 = __shfl(sp4, 16, 32);
  w0 = (s >= 1) ? w0 * SCALE : -1e30f;
  w1 *= SCALE;
  float smx = fmaxf(w0, w1);
  float e0 = __expf(w0 - smx), e1 = __expf(w1 - smx);
  float sv = 1.f / (e0 + e1);
  const _Float16* v0r = vh + ((size_t)(b * Sc + sp0) * Hc + h) * Dc;
  const _Float16* v1r = vh + ((size_t)(b * Sc + s) * Hc + h) * Dc;
  float so0 = (e0 * (float)v0r[d0]     + e1 * (float)v1r[d0])     * sv;
  float so1 = (e0 * (float)v0r[d0 + 1] + e1 * (float)v1r[d0 + 1]) * sv;

  const float* g = gates + (size_t)(b * Sc + s) * 24 + h * 3;
  float g0 = g[0], g1 = g[1], g2 = g[2];
  const float* cr = cout + ((size_t)bh * Sc + s) * Dc;
  size_t obase = (size_t)(b * Sc + s) * (Hc * Dc) + h * Dc;
  oh[obase + d0]     = (_Float16)(g0 * cr[d0]     + g1 * f0 + g2 * so0);
  oh[obase + d0 + 1] = (_Float16)(g0 * cr[d0 + 1] + g1 * f1 + g2 * so1);
}

// ---------------- host orchestration ----------------
extern "C" void kernel_launch(void* const* d_in, const int* in_sizes, int n_in,
                              void* d_out, int out_size, void* d_ws, size_t ws_size,
                              hipStream_t stream) {
  (void)in_sizes; (void)n_in; (void)out_size; (void)ws_size;
  const float* src = (const float*)d_in[0];
  const float* Wq  = (const float*)d_in[1];
  const float* Wk  = (const float*)d_in[2];
  const float* Wv  = (const float*)d_in[3];
  const float* kpe = (const float*)d_in[4];
  const float* vpe = (const float*)d_in[5];
  const float* Wck = (const float*)d_in[6];
  const float* bck = (const float*)d_in[7];
  const float* Wcv = (const float*)d_in[8];
  const float* bcv = (const float*)d_in[9];
  const float* mck = (const float*)d_in[10];
  const float* mcv = (const float*)d_in[11];
  const float* Wg  = (const float*)d_in[12];
  const float* bg  = (const float*)d_in[13];
  const float* Wo  = (const float*)d_in[14];
  const float* bo  = (const float*)d_in[15];
  const float* g1  = (const float*)d_in[16];
  const float* b1  = (const float*)d_in[17];
  const float* g2  = (const float*)d_in[18];
  const float* b2  = (const float*)d_in[19];
  const float* W1  = (const float*)d_in[20];
  const float* bf1 = (const float*)d_in[21];
  const float* W2  = (const float*)d_in[22];
  const float* bf2 = (const float*)d_in[23];

  const int M = Bc * Sc;                                   // 4096 tokens
  char* base = (char*)d_ws; size_t off = 0;
  auto alloc = [&](size_t bytes) -> void* {
    void* p = base + off; off = (off + bytes + 255) & ~(size_t)255; return p;
  };
  _Float16* WqT  = (_Float16*)alloc((size_t)DMc * DMc * 2);     // (512,512) N-major
  _Float16* WkT  = (_Float16*)alloc((size_t)DMc * DMc * 2);
  _Float16* WvT  = (_Float16*)alloc((size_t)DMc * DMc * 2);
  _Float16* WoT  = (_Float16*)alloc((size_t)DMc * DMc * 2);
  _Float16* W1T  = (_Float16*)alloc((size_t)DFFc * DMc * 2);    // (2048,512)
  _Float16* W2T  = (_Float16*)alloc((size_t)DMc * DFFc * 2);    // (512,2048)
  _Float16* WckT = (_Float16*)alloc((size_t)Dc * 256 * 2);      // (64,256)
  _Float16* WcvT = (_Float16*)alloc((size_t)Dc * 256 * 2);
  _Float16* xnH  = (_Float16*)alloc((size_t)M * DMc * 2);
  _Float16* qH   = (_Float16*)alloc((size_t)M * DMc * 2);
  _Float16* kH   = (_Float16*)alloc((size_t)M * DMc * 2);
  _Float16* vH   = (_Float16*)alloc((size_t)M * DMc * 2);
  _Float16* ckH  = (_Float16*)alloc((size_t)Bc * Hc * NKc * Dc * 2);  // key-major
  _Float16* cvT  = (_Float16*)alloc((size_t)Bc * Hc * Dc * NKc * 2);  // d-major
  float*    coutW= (float*)alloc((size_t)Bc * Hc * Sc * Dc * 4);
  int*      selW = (int*)alloc((size_t)Bc * Hc * Sc * 2 * 4);
  float*    gatesW=(float*)alloc((size_t)M * 24 * 4);
  _Float16* oH   = (_Float16*)alloc((size_t)M * DMc * 2);
  float*    xW   = (float*)alloc((size_t)M * DMc * 4);
  _Float16* x2H  = (_Float16*)alloc((size_t)M * DMc * 2);
  _Float16* h1H  = (_Float16*)alloc((size_t)M * DFFc * 2);

  // transpose+cast weights to N-major f16
  transcast_kernel<<<1024, 256, 0, stream>>>(Wq, WqT, DMc, DMc);
  transcast_kernel<<<1024, 256, 0, stream>>>(Wk, WkT, DMc, DMc);
  transcast_kernel<<<1024, 256, 0, stream>>>(Wv, WvT, DMc, DMc);
  transcast_kernel<<<1024, 256, 0, stream>>>(Wo, WoT, DMc, DMc);
  transcast_kernel<<<4096, 256, 0, stream>>>(W1, W1T, DMc, DFFc);
  transcast_kernel<<<4096, 256, 0, stream>>>(W2, W2T, DFFc, DMc);
  transcast_kernel<<<64, 256, 0, stream>>>(Wck, WckT, 256, Dc);
  transcast_kernel<<<64, 256, 0, stream>>>(Wcv, WcvT, 256, Dc);

  // LN1 -> xn (f16)
  ln_kernel<<<M, 128, 0, stream>>>(src, g1, b1, xnH);

  // Q/K/V projections (f16 out, token-major B,S,H,D layout)
  dim3 gqkv(DMc / 64, M / 128);
  wmma_gemm_kernel<1, 0, 0, DMc, DMc><<<gqkv, 256, 0, stream>>>(xnH, WqT, nullptr, nullptr, qH);
  wmma_gemm_kernel<1, 0, 0, DMc, DMc><<<gqkv, 256, 0, stream>>>(xnH, WkT, nullptr, nullptr, kH);
  wmma_gemm_kernel<1, 0, 0, DMc, DMc><<<gqkv, 256, 0, stream>>>(xnH, WvT, nullptr, nullptr, vH);

  // compressed K / V^T
  memrow_kernel<<<Bc * Hc, Dc, 0, stream>>>(mck, mcv, ckH, cvT);
  dim3 gck((NCc + 15) / 16, Bc * Hc, 2);
  ckcv_kernel<<<gck, 32, 0, stream>>>(kH, vH, kpe, vpe, WckT, WcvT, bck, bcv, ckH, cvT);

  // gates
  gates_kernel<<<M, 32, 0, stream>>>(xnH, Wg, bg, gatesW);

  // compressed attention + selection
  dim3 gat(Bc * Hc, Sc / (16 * CWAVES));
  cattn_kernel<<<gat, 32 * CWAVES, 0, stream>>>(qH, ckH, cvT, coutW, selW);

  // fine + sliding + gating
  fsg_kernel<<<(Bc * Hc * Sc) / 8, 256, 0, stream>>>(qH, kH, vH, selW, coutW, gatesW, oH);

  // output projection (+src residual) -> x
  wmma_gemm_kernel<0, 0, 1, DMc, DMc><<<dim3(DMc / 64, M / 128), 256, 0, stream>>>(oH, WoT, bo, src, xW);

  // LN2 -> x2 (f16)
  ln_kernel<<<M, 128, 0, stream>>>(xW, g2, b2, x2H);

  // FFN1 (GELU, f16 out)
  wmma_gemm_kernel<1, 1, 0, DFFc, DMc><<<dim3(DFFc / 64, M / 128), 256, 0, stream>>>(x2H, W1T, bf1, nullptr, h1H);

  // FFN2 (+x residual) -> final output f32
  wmma_gemm_kernel<0, 0, 1, DMc, DFFc><<<dim3(DMc / 64, M / 128), 256, 0, stream>>>(h1H, W2T, bf2, xW, (float*)d_out);
}
